// GAE_63273458204921
// MI455X (gfx1250) — matrix operand order repaired and checked by reading, not compiled, not run on hardware
//
#include <hip/hip_runtime.h>
#include <cstdint>

typedef __attribute__((ext_vector_type(16))) _Float16 v16h;
typedef __attribute__((ext_vector_type(8)))  float    v8f;

// ---------------- degree / normalization ----------------

__global__ void GCN_deg_init(float* __restrict__ deg, int n) {
  int i = blockIdx.x * blockDim.x + threadIdx.x;
  if (i < n) deg[i] = 1.0f;  // self-loop weight contributes 1 to every node
}

__global__ void GCN_deg_accum(const int* __restrict__ ei, const float* __restrict__ ew,
                              float* __restrict__ deg, int E) {
  int e = blockIdx.x * blockDim.x + threadIdx.x;
  if (e < E) atomicAdd(&deg[ei[E + e]], ew[e]);  // dst = ei[1][e]
}

__global__ void GCN_dinv(float* __restrict__ deg, int n) {
  int i = blockIdx.x * blockDim.x + threadIdx.x;
  if (i < n) {
    float d = deg[i];
    deg[i] = (d > 0.0f) ? rsqrtf(d) : 0.0f;  // in place: deg -> dinv
  }
}

// ---------------- WMMA GEMM: C[M,NCOLS] = A[M,K] * B[K,NCOLS] ----------------
// f32 inputs converted to f16 fragments, f32 accumulation via
// v_wmma_f32_16x16x32_f16. One 16x16 C tile per wave32. K % 32 == 0.

template <int K, int NCOLS>
__global__ void GCN_gemm_wmma(const float* __restrict__ A, const float* __restrict__ B,
                              float* __restrict__ C, int mtiles) {
  const int ntiles = NCOLS / 16;
  int wave = blockIdx.x * (blockDim.x >> 5) + (threadIdx.x >> 5);
  int tm = wave / ntiles;
  if (tm >= mtiles) return;            // uniform per wave -> EXEC all-ones for WMMA
  int tn   = wave % ntiles;
  int lane = threadIdx.x & 31;
  int half = lane >> 4;                // 0: lanes 0-15, 1: lanes 16-31
  int lh   = lane & 15;

  const float* arow = A + (size_t)(tm * 16 + lh) * K;   // A row for this lane
  const float* bcol = B + (size_t)(tn * 16 + lh);       // B column for this lane
  const int kA0 = half * 16;       // elements 0..7  <-> K = kA0..kA0+7
  const int kA1 = half * 16 + 8;   // elements 8..15 <-> K = kA1..kA1+7
  const int kB  = half * 16;       // B elements 0..15 <-> K = kB..kB+15

  v8f acc = {};
  for (int kk = 0; kk < K; kk += 32) {
    float af[16];
    *(float4*)(af + 0)  = *(const float4*)(arow + kk + kA0);
    *(float4*)(af + 4)  = *(const float4*)(arow + kk + kA0 + 4);
    *(float4*)(af + 8)  = *(const float4*)(arow + kk + kA1);
    *(float4*)(af + 12) = *(const float4*)(arow + kk + kA1 + 4);
    v16h a, b;
#pragma unroll
    for (int i = 0; i < 16; ++i) a[i] = (_Float16)af[i];
#pragma unroll
    for (int i = 0; i < 16; ++i) b[i] = (_Float16)bcol[(size_t)(kk + kB + i) * NCOLS];
    acc = __builtin_amdgcn_wmma_f32_16x16x32_f16(
        /*neg_a=*/false, a, /*neg_b=*/false, b,
        /*c_mod=*/(short)0, acc, /*reuse_a=*/false, /*reuse_b=*/false);
  }

  float* crow = C + (size_t)(tm * 16 + half * 8) * NCOLS + tn * 16 + lh;
#pragma unroll
  for (int r = 0; r < 8; ++r) crow[(size_t)r * NCOLS] = acc[r];
}

// ---------------- self-loop init: out = dinv[i]^2 * h + (bias?) ----------------

__global__ void GCN_self_init(const float* __restrict__ h, const float* __restrict__ dinv,
                              const float* __restrict__ bias, float* __restrict__ out,
                              int total, int cshift, int cmask, int add_bias) {
  int idx = blockIdx.x * blockDim.x + threadIdx.x;
  if (idx >= total) return;
  int i = idx >> cshift;
  int c = idx & cmask;
  float v = dinv[i];
  out[idx] = v * v * h[idx] + (add_bias ? bias[c] : 0.0f);
}

// ---------------- edge gather/scale/scatter-add: one wave32 per edge ----------------

__global__ void GCN_edge_scatter(const float* __restrict__ h, const int* __restrict__ ei,
                                 const float* __restrict__ ew, const float* __restrict__ dinv,
                                 float* __restrict__ out, int E, int C) {
  int e = blockIdx.x * (blockDim.x >> 5) + (threadIdx.x >> 5);
  if (e >= E) return;
  int lane = threadIdx.x & 31;
  int s = ei[e];
  int d = ei[E + e];
  float norm = dinv[s] * ew[e] * dinv[d];
  const float* hs = h + (size_t)s * C;
  float* od = out + (size_t)d * C;
  for (int c = lane; c < C; c += 32)
    atomicAdd(&od[c], hs[c] * norm);
}

// ---------------- relu(x + b) in place ----------------

__global__ void GCN_relu_bias(float* __restrict__ a, const float* __restrict__ b,
                              int total, int cmask) {
  int idx = blockIdx.x * blockDim.x + threadIdx.x;
  if (idx < total) a[idx] = fmaxf(a[idx] + b[idx & cmask], 0.0f);
}

// ---------------- launch ----------------

static inline size_t align256(size_t x) { return (x + 255) & ~(size_t)255; }

extern "C" void kernel_launch(void* const* d_in, const int* in_sizes, int n_in,
                              void* d_out, int out_size, void* d_ws, size_t ws_size,
                              hipStream_t stream) {
  (void)n_in; (void)out_size; (void)ws_size;
  const float* x  = (const float*)d_in[0];
  const int*   ei = (const int*)d_in[1];
  const float* ew = (const float*)d_in[2];
  const float* W1 = (const float*)d_in[3];
  const float* b1 = (const float*)d_in[4];
  const float* W2 = (const float*)d_in[5];
  const float* b2 = (const float*)d_in[6];
  float* out = (float*)d_out;

  const int FDIM = 512, H = 256, EMB = 128;
  const int N = in_sizes[0] / FDIM;   // 50000
  const int E = in_sizes[2];          // 1600000

  // workspace partition
  char* ws = (char*)d_ws;
  size_t off = 0;
  float* dinv = (float*)(ws + off); off += align256((size_t)N * sizeof(float));
  float* h    = (float*)(ws + off); off += align256((size_t)N * H * sizeof(float));
  float* h1   = (float*)(ws + off); off += align256((size_t)N * H * sizeof(float));
  float* h2   = (float*)(ws + off); off += align256((size_t)N * EMB * sizeof(float));

  const int T = 256;                 // 8 waves per block
  const int wavesPerBlk = T / 32;

  // 1) degree with self-loops, then dinv in place
  GCN_deg_init<<<(N + T - 1) / T, T, 0, stream>>>(dinv, N);
  GCN_deg_accum<<<(E + T - 1) / T, T, 0, stream>>>(ei, ew, dinv, E);
  GCN_dinv<<<(N + T - 1) / T, T, 0, stream>>>(dinv, N);

  // 2) h = x @ W1  (WMMA)
  {
    int mtiles = N / 16;                         // 3125
    int waves  = mtiles * (H / 16);              // 50000
    GCN_gemm_wmma<512, 256><<<(waves + wavesPerBlk - 1) / wavesPerBlk, T, 0, stream>>>(
        x, W1, h, mtiles);
  }

  // 3) layer-1 aggregation: self loop init, edge scatter, relu(+b1)
  {
    int total = N * H;
    GCN_self_init<<<(total + T - 1) / T, T, 0, stream>>>(h, dinv, b1, h1, total, 8, 255, 0);
    GCN_edge_scatter<<<(E + wavesPerBlk - 1) / wavesPerBlk, T, 0, stream>>>(
        h, ei, ew, dinv, h1, E, H);
    GCN_relu_bias<<<(total + T - 1) / T, T, 0, stream>>>(h1, b1, total, 255);
  }

  // 4) h2 = h1 @ W2  (WMMA)
  {
    int mtiles = N / 16;
    int waves  = mtiles * (EMB / 16);            // 25000
    GCN_gemm_wmma<256, 128><<<(waves + wavesPerBlk - 1) / wavesPerBlk, T, 0, stream>>>(
        h1, W2, h2, mtiles);
  }

  // 5) layer-2 aggregation straight into d_out: self loop + b2 init, edge scatter
  {
    int total = N * EMB;
    GCN_self_init<<<(total + T - 1) / T, T, 0, stream>>>(h2, dinv, b2, out, total, 7, 127, 1);
    GCN_edge_scatter<<<(E + wavesPerBlk - 1) / wavesPerBlk, T, 0, stream>>>(
        h2, ei, ew, dinv, out, E, EMB);
  }
}
